// MusicInformer_9423158247871
// MI455X (gfx1250) — matrix operand order repaired and checked by reading, not compiled, not run on hardware
//
#include <hip/hip_runtime.h>
#include <cmath>

typedef unsigned short u16;
typedef unsigned int   u32;

constexpr int NB = 4, SEQLEN = 2048, DMODEL = 512, DFFN = 1024, NHEAD = 8, HDIM = 64, NVOCAB = 390;

typedef __attribute__((ext_vector_type(16))) __bf16 bf16x16;
typedef __attribute__((ext_vector_type(8)))  float  f32x8;

// ---- CDNA5 async global->LDS path (ASYNCcnt), with graceful fallback
#if defined(__has_builtin)
#if __has_builtin(__builtin_amdgcn_global_load_async_to_lds_b128) && __has_builtin(__builtin_amdgcn_s_wait_asynccnt)
#define USE_ASYNC_LDS 1
#endif
#endif
#ifndef USE_ASYNC_LDS
#define USE_ASYNC_LDS 0
#endif
// builtin signature (from hipcc diagnostic): (v4i AS1* src, v4i AS3* dst, imm int offset, imm int cpol)
typedef int v4i __attribute__((vector_size(16)));
typedef __attribute__((address_space(1))) v4i* as1_v4i_p;
typedef __attribute__((address_space(3))) v4i* as3_v4i_p;

__device__ __forceinline__ float b2f(u16 h) { union { u32 u; float f; } x; x.u = ((u32)h) << 16; return x.f; }
__device__ __forceinline__ u16 f2b(float f) {
    union { float f; u32 u; } x; x.f = f;
    u32 r = x.u + 0x7fffu + ((x.u >> 16) & 1u);
    return (u16)(r >> 16);
}
__device__ __forceinline__ float sigm(float x) { return 1.0f / (1.0f + expf(-x)); }

// ---------------------------------------------------------------- utility kernels
__global__ void k_fill0(float* p, int n) {
    int i = blockIdx.x * blockDim.x + threadIdx.x;
    if (i < n) p[i] = 0.0f;
}
__global__ void k_cvt(const float* __restrict__ s, u16* __restrict__ d, int n) {
    int i = blockIdx.x * blockDim.x + threadIdx.x;
    if (i < n) d[i] = f2b(s[i]);
}

// embedding * sqrt(D_FF) + sinusoid PE; writes f32 + bf16 copies
__global__ void k_embed(const int* __restrict__ x, const float* __restrict__ emb,
                        float* __restrict__ hf, u16* __restrict__ hb) {
    int idx = blockIdx.x * blockDim.x + threadIdx.x;
    if (idx >= NB * SEQLEN * DMODEL) return;
    int d = idx % DMODEL;
    int l = (idx / DMODEL) % SEQLEN;
    int bl = idx / DMODEL;           // b*SEQLEN + l
    int tok = x[bl];
    float v = emb[(size_t)tok * DMODEL + d] * 32.0f;   // sqrt(D_FF) = 32
    int i2 = d & ~1;
    float div = expf(-(float)i2 * (9.210340372f / (float)DMODEL));
    float ang = (float)l * div;
    v += (d & 1) ? cosf(ang) : sinf(ang);
    hf[idx] = v;
    hb[idx] = f2b(v);
}

// ---------------------------------------------------------------- WMMA GEMM
// C[M,N] = act( A[M,K](bf16) * W[N,K]^T(bf16) + bias[N] )
// tile 64(M) x 128(N) x 32(K), 8 waves (2x4), each wave: 2x2 frags of 16x16
// double-buffered LDS; unguarded instantiation uses async global->LDS DMA
#define TM 64
#define TN 128
#define TK 32
#define LPAD 8
union ABfrag { uint4 q[2]; bf16x16 v; };

template <bool GUARD>
__global__ __launch_bounds__(256) void k_gemm(const u16* __restrict__ A, const u16* __restrict__ W,
                                              const float* __restrict__ bias,
                                              float* __restrict__ outF, u16* __restrict__ outB,
                                              int M, int N, int K, int relu) {
    __shared__ __align__(16) u16 As[2][TM][TK + LPAD];
    __shared__ __align__(16) u16 Bs[2][TN][TK + LPAD];
    int tid  = threadIdx.x;
    int lane = tid & 31, wid = tid >> 5;
    int wm = wid >> 2, wn = wid & 3;                 // 2 x 4 wave grid
    int m0 = blockIdx.y * TM, n0 = blockIdx.x * TN;
    int arow = tid >> 2, ach = tid & 3;              // A staging: 64 rows x 4 chunks
    int brow0 = tid >> 2, brow1 = (tid + 256) >> 2;  // B staging: 128 rows x 4 chunks, 2/thread
    int nkt = K / TK;

    f32x8 acc[2][2];
    for (int i = 0; i < 2; i++)
        for (int j = 0; j < 2; j++) acc[i][j] = (f32x8)(0.0f);

    // fragment build + 4 WMMAs on buffer p (ISA 7.12.2 layouts)
    auto compute = [&](int p) {
        int r = lane & 15;
        int aoff = (lane >> 4) * 8;     // A: upper lane-half starts at K=8, chunks at +0 and +16
        int boff = (lane >> 4) * 8;     // B: upper lane-half starts at K=8, chunks at +0 and +16
        ABfrag a[2], b[2];
        for (int i = 0; i < 2; i++) {
            int ar = wm * 32 + i * 16 + r;
            a[i].q[0] = *(const uint4*)&As[p][ar][aoff];
            a[i].q[1] = *(const uint4*)&As[p][ar][aoff + 16];
        }
        for (int j = 0; j < 2; j++) {
            int br = wn * 32 + j * 16 + r;
            b[j].q[0] = *(const uint4*)&Bs[p][br][boff];
            b[j].q[1] = *(const uint4*)&Bs[p][br][boff + 16];
        }
        for (int i = 0; i < 2; i++)
            for (int j = 0; j < 2; j++)
                acc[i][j] = __builtin_amdgcn_wmma_f32_16x16x32_bf16(
                    false, a[i].v, false, b[j].v, (short)0, acc[i][j], false, false);
    };

#if USE_ASYNC_LDS
    if constexpr (!GUARD) {
        // async DMA pipeline: issue tile t+1 while computing tile t
        auto issue = [&](int p, int k0) {
            __builtin_amdgcn_global_load_async_to_lds_b128(
                (as1_v4i_p)(A + (size_t)(m0 + arow) * K + k0 + ach * 8),
                (as3_v4i_p)&As[p][arow][ach * 8], 0, 0);
            __builtin_amdgcn_global_load_async_to_lds_b128(
                (as1_v4i_p)(W + (size_t)(n0 + brow0) * K + k0 + ach * 8),
                (as3_v4i_p)&Bs[p][brow0][ach * 8], 0, 0);
            __builtin_amdgcn_global_load_async_to_lds_b128(
                (as1_v4i_p)(W + (size_t)(n0 + brow1) * K + k0 + ach * 8),
                (as3_v4i_p)&Bs[p][brow1][ach * 8], 0, 0);
        };
        issue(0, 0);
        for (int t = 0; t < nkt; t++) {
            int p = t & 1;
            if (t + 1 < nkt) {
                issue(1 - p, (t + 1) * TK);           // overwrite of buf consumed last iter (safe: barrier below)
                __builtin_amdgcn_s_wait_asynccnt(3);  // current tile's 3 DMAs complete; next tile in flight
            } else {
                __builtin_amdgcn_s_wait_asynccnt(0);
            }
            __syncthreads();                          // all waves' chunks of buf p landed
            compute(p);
            __syncthreads();                          // everyone done reading buf p before it is re-DMA'd
        }
    } else
#endif
    {
        // software-pipelined sync path (also handles edge tiles)
        const uint4 zero4 = {0u, 0u, 0u, 0u};
        uint4 ra, rb0, rb1;
        auto gload = [&](int k0) {
            ra = zero4; rb0 = zero4; rb1 = zero4;
            if (!GUARD || (m0 + arow) < M)  ra  = *(const uint4*)(A + (size_t)(m0 + arow)  * K + k0 + ach * 8);
            if (!GUARD || (n0 + brow0) < N) rb0 = *(const uint4*)(W + (size_t)(n0 + brow0) * K + k0 + ach * 8);
            if (!GUARD || (n0 + brow1) < N) rb1 = *(const uint4*)(W + (size_t)(n0 + brow1) * K + k0 + ach * 8);
        };
        auto stash = [&](int p) {
            *(uint4*)&As[p][arow][ach * 8]  = ra;
            *(uint4*)&Bs[p][brow0][ach * 8] = rb0;
            *(uint4*)&Bs[p][brow1][ach * 8] = rb1;
        };
        gload(0);
        stash(0);
        for (int t = 0; t < nkt; t++) {
            int p = t & 1;
            bool more = (t + 1) < nkt;
            if (more) {
                gload((t + 1) * TK);                  // overlaps with compute(p) below
                __builtin_prefetch(A + (size_t)(m0 + arow) * K + (t + 1) * TK + ach * 8, 0, 1);
            }
            __syncthreads();                          // buf p stores visible
            compute(p);
            __syncthreads();                          // buf 1-p free to overwrite
            if (more) stash(1 - p);
        }
    }

    // store: C/D layout — VGPR e: M = e (+8 for upper lane-half), N = lane%16
    int rhalf = (lane >> 4) * 8;
    int cn = lane & 15;
    for (int i = 0; i < 2; i++)
        for (int j = 0; j < 2; j++)
            for (int e = 0; e < 8; e++) {
                int m = m0 + wm * 32 + i * 16 + e + rhalf;
                int n = n0 + wn * 32 + j * 16 + cn;
                if (!GUARD || (m < M && n < N)) {
                    float v = acc[i][j][e];
                    if (bias) v += bias[n];
                    if (relu) v = fmaxf(v, 0.0f);
                    if (outF) outF[(size_t)m * N + n] = v;
                    if (outB) outB[(size_t)m * N + n] = f2b(v);
                }
            }
}

// ---------------------------------------------------------------- LayerNorm (+optional residual)
__global__ __launch_bounds__(256) void k_ln(const float* __restrict__ x, const float* __restrict__ res,
                                            const float* __restrict__ g, const float* __restrict__ bt,
                                            float* __restrict__ outF, u16* __restrict__ outB, int rows) {
    int row = blockIdx.x;
    if (row >= rows) return;
    int tid = threadIdx.x;
    const float* xr = x + (size_t)row * DMODEL;
    const float* rr = res ? res + (size_t)row * DMODEL : nullptr;
    float v0 = xr[tid]       + (rr ? rr[tid]       : 0.0f);
    float v1 = xr[tid + 256] + (rr ? rr[tid + 256] : 0.0f);
    __shared__ float red[256];
    red[tid] = v0 + v1;
    __syncthreads();
    for (int s = 128; s > 0; s >>= 1) { if (tid < s) red[tid] += red[tid + s]; __syncthreads(); }
    float mean = red[0] * (1.0f / DMODEL);
    __syncthreads();
    float d0 = v0 - mean, d1 = v1 - mean;
    red[tid] = d0 * d0 + d1 * d1;
    __syncthreads();
    for (int s = 128; s > 0; s >>= 1) { if (tid < s) red[tid] += red[tid + s]; __syncthreads(); }
    float rstd = rsqrtf(red[0] * (1.0f / DMODEL) + 1e-5f);
    float o0 = d0 * rstd * g[tid]       + bt[tid];
    float o1 = d1 * rstd * g[tid + 256] + bt[tid + 256];
    size_t o = (size_t)row * DMODEL;
    if (outF) { outF[o + tid] = o0; outF[o + tid + 256] = o1; }
    if (outB) { outB[o + tid] = f2b(o0); outB[o + tid + 256] = f2b(o1); }
}

// ---------------------------------------------------------------- ProbAttention pieces
// sparsity measure: M[l] = max_j qk_s - sum_j qk_s / L  over u pseudo-random sampled keys
__global__ void k_probM(const u16* __restrict__ Q, const u16* __restrict__ K,
                        float* __restrict__ Mo, int L, int u) {
    int idx = blockIdx.x * blockDim.x + threadIdx.x;
    if (idx >= NB * NHEAD * L) return;
    int l = idx % L;
    int bh = idx / L;
    int h = bh % NHEAD, b = bh / NHEAD;
    size_t base = ((size_t)b * L) * DMODEL + h * HDIM;
    const u16* q = Q + base + (size_t)l * DMODEL;
    float qv[HDIM];
#pragma unroll
    for (int d = 0; d < HDIM; d++) qv[d] = b2f(q[d]);
    float mx = -3.4e38f, sm = 0.0f;
    for (int j = 0; j < u; j++) {
        u32 kidx = (2654435761u * (u32)(l + 1) + 40503u * (u32)(j + 1)) % (u32)L;
        const u16* kp = K + base + (size_t)kidx * DMODEL;
        float s = 0.0f;
#pragma unroll
        for (int d = 0; d < HDIM; d++) s += qv[d] * b2f(kp[d]);
        mx = fmaxf(mx, s);
        sm += s;
    }
    Mo[idx] = mx - sm / (float)L;
}

// top-u selection per (b,h): iterative masked argmax in LDS
__global__ __launch_bounds__(256) void k_topu(const float* __restrict__ Min, int* __restrict__ mtop,
                                              int L, int u) {
    int bh = blockIdx.x, tid = threadIdx.x;
    __shared__ float buf[2048];
    __shared__ float sv[256];
    __shared__ int   si[256];
    const float* src = Min + (size_t)bh * L;
    for (int i = tid; i < L; i += 256) buf[i] = src[i];
    __syncthreads();
    for (int it = 0; it < u; ++it) {
        float best = -3.4e38f; int bi = 0;
        for (int i = tid; i < L; i += 256) { float v = buf[i]; if (v > best) { best = v; bi = i; } }
        sv[tid] = best; si[tid] = bi;
        __syncthreads();
        for (int s = 128; s > 0; s >>= 1) {
            if (tid < s && sv[tid + s] > sv[tid]) { sv[tid] = sv[tid + s]; si[tid] = si[tid + s]; }
            __syncthreads();
        }
        if (tid == 0) { mtop[bh * u + it] = si[0]; buf[si[0]] = -3.4e38f; }
        __syncthreads();
    }
}

// lazy-query context: ctx[b,l,h,:] = mean_l V ; block per (b,h), 64 threads
__global__ void k_vmean(const u16* __restrict__ V, u16* __restrict__ CTX, int L) {
    int bh = blockIdx.x;
    int h = bh % NHEAD, b = bh / NHEAD;
    int d = threadIdx.x;
    size_t base = ((size_t)b * L) * DMODEL + h * HDIM;
    float s = 0.0f;
    for (int l = 0; l < L; l++) s += b2f(V[base + (size_t)l * DMODEL + d]);
    u16 sb = f2b(s / (float)L);
    for (int l = 0; l < L; l++) CTX[base + (size_t)l * DMODEL + d] = sb;
}

// full softmax rows for the top-u queries, scatter into ctx
__global__ __launch_bounds__(256) void k_toprows(const u16* __restrict__ Q, const u16* __restrict__ K,
                                                 const u16* __restrict__ V, const int* __restrict__ mtop,
                                                 u16* __restrict__ CTX, int L, int u) {
    int gid = blockIdx.x;
    int j = gid % u, bh = gid / u;
    int h = bh % NHEAD, b = bh / NHEAD;
    int tid = threadIdx.x;
    __shared__ float qs[HDIM];
    __shared__ float sc[2048];
    __shared__ float red[256];
    size_t base = ((size_t)b * L) * DMODEL + h * HDIM;
    int qi = mtop[bh * u + j];
    if (tid < HDIM) qs[tid] = b2f(Q[base + (size_t)qi * DMODEL + tid]);
    __syncthreads();
    float mx = -3.4e38f;
    for (int l = tid; l < L; l += 256) {
        const u16* kp = K + base + (size_t)l * DMODEL;
        float s = 0.0f;
#pragma unroll
        for (int d = 0; d < HDIM; d++) s += qs[d] * b2f(kp[d]);
        s *= 0.125f;                     // 1/sqrt(64)
        sc[l] = s;
        mx = fmaxf(mx, s);
    }
    red[tid] = mx; __syncthreads();
    for (int s = 128; s > 0; s >>= 1) { if (tid < s) red[tid] = fmaxf(red[tid], red[tid + s]); __syncthreads(); }
    float M = red[0]; __syncthreads();
    float sum = 0.0f;
    for (int l = tid; l < L; l += 256) { float e = expf(sc[l] - M); sc[l] = e; sum += e; }
    red[tid] = sum; __syncthreads();
    for (int s = 128; s > 0; s >>= 1) { if (tid < s) red[tid] += red[tid + s]; __syncthreads(); }
    float inv = 1.0f / red[0];
    if (tid < HDIM) {
        int d = tid;
        float a = 0.0f;
        for (int l = 0; l < L; l++) a += sc[l] * b2f(V[base + (size_t)l * DMODEL + d]);
        CTX[base + (size_t)qi * DMODEL + d] = f2b(a * inv);
    }
}

// standard causal attention for the RPR layers (L<=256)
__global__ __launch_bounds__(256) void k_causal(const u16* __restrict__ Q, const u16* __restrict__ K,
                                                const u16* __restrict__ V, u16* __restrict__ CTX, int L) {
    int gid = blockIdx.x;
    int lq = gid % L, bh = gid / L;
    int h = bh % NHEAD, b = bh / NHEAD;
    int tid = threadIdx.x;
    __shared__ float qs[HDIM];
    __shared__ float sc[256];
    __shared__ float red[256];
    size_t base = ((size_t)b * L) * DMODEL + h * HDIM;
    if (tid < HDIM) qs[tid] = b2f(Q[base + (size_t)lq * DMODEL + tid]);
    __syncthreads();
    int Lk = lq + 1;
    float mx = -3.4e38f;
    for (int l = tid; l < Lk; l += 256) {
        const u16* kp = K + base + (size_t)l * DMODEL;
        float s = 0.0f;
#pragma unroll
        for (int d = 0; d < HDIM; d++) s += qs[d] * b2f(kp[d]);
        s *= 0.125f;
        sc[l] = s;
        mx = fmaxf(mx, s);
    }
    red[tid] = mx; __syncthreads();
    for (int s = 128; s > 0; s >>= 1) { if (tid < s) red[tid] = fmaxf(red[tid], red[tid + s]); __syncthreads(); }
    float M = red[0]; __syncthreads();
    float sum = 0.0f;
    for (int l = tid; l < Lk; l += 256) { float e = expf(sc[l] - M); sc[l] = e; sum += e; }
    red[tid] = sum; __syncthreads();
    for (int s = 128; s > 0; s >>= 1) { if (tid < s) red[tid] += red[tid + s]; __syncthreads(); }
    float inv = 1.0f / red[0];
    if (tid < HDIM) {
        int d = tid;
        float a = 0.0f;
        for (int l = 0; l < Lk; l++) a += sc[l] * b2f(V[base + (size_t)l * DMODEL + d]);
        CTX[base + (size_t)lq * DMODEL + d] = f2b(a * inv);
    }
}

// ---------------------------------------------------------------- conv-distill (im2col + bn/elu/pool)
__global__ void k_im2col(const u16* __restrict__ x, u16* __restrict__ xc, int L) {
    int idx = blockIdx.x * blockDim.x + threadIdx.x;
    if (idx >= NB * L * DMODEL) return;
    int ci = idx % DMODEL;
    int l = (idx / DMODEL) % L;
    int b = idx / (DMODEL * L);
    size_t orow = ((size_t)b * L + l) * (3 * DMODEL) + ci * 3;
    for (int dk = 0; dk < 3; dk++) {
        int li = l + dk - 1;
        if (li < 0) li += L;
        if (li >= L) li -= L;
        xc[orow + dk] = x[((size_t)b * L + li) * DMODEL + ci];
    }
}
__global__ void k_pool(const float* __restrict__ conv, const float* __restrict__ g, const float* __restrict__ bb,
                       float* __restrict__ hf, u16* __restrict__ hb, int Lin, int Lout) {
    int idx = blockIdx.x * blockDim.x + threadIdx.x;
    if (idx >= NB * Lout * DMODEL) return;
    int c = idx % DMODEL;
    int lo = (idx / DMODEL) % Lout;
    int b = idx / (DMODEL * Lout);
    float m = -3.4e38f;
    for (int dk = 0; dk < 3; dk++) {
        int li = 2 * lo - 1 + dk;
        if (li < 0 || li >= Lin) continue;
        float v = conv[((size_t)b * Lin + li) * DMODEL + c] * g[c] + bb[c];
        v = v > 0.0f ? v : (expf(v) - 1.0f);   // ELU
        m = fmaxf(m, v);
    }
    size_t o = ((size_t)b * Lout + lo) * DMODEL + c;
    hf[o] = m;
    hb[o] = f2b(m);
}

// ---------------------------------------------------------------- LSTM step (wave per (b,j))
__global__ __launch_bounds__(256) void k_lstm_step(const float* __restrict__ x, const u16* __restrict__ wih,
                                                   const u16* __restrict__ whh, const float* __restrict__ bih,
                                                   const float* __restrict__ bhh,
                                                   const float* __restrict__ hin, const float* __restrict__ cin,
                                                   float* __restrict__ hout, float* __restrict__ cout,
                                                   float* __restrict__ ys, int t, int Kin) {
    int lane = threadIdx.x & 31;
    int gw = blockIdx.x * 8 + (threadIdx.x >> 5);    // over NB*DFFN
    int j = gw % DFFN, b = gw / DFFN;
    const float* xp = x + ((size_t)b * 64 + t) * Kin;
    const float* hp = hin + (size_t)b * DFFN;
    float g4[4];
#pragma unroll
    for (int gi = 0; gi < 4; gi++) {
        int row = gi * DFFN + j;
        const u16* wi = wih + (size_t)row * Kin;
        const u16* wh = whh + (size_t)row * DFFN;
        float s = 0.0f;
        for (int k = lane; k < Kin; k += 32) s += xp[k] * b2f(wi[k]);
        for (int k = lane; k < DFFN; k += 32) s += hp[k] * b2f(wh[k]);
        for (int off = 16; off > 0; off >>= 1) s += __shfl_down(s, off, 32);
        g4[gi] = s;
    }
    if (lane == 0) {
        float gi = sigm(g4[0] + bih[j] + bhh[j]);
        float gf = sigm(g4[1] + bih[DFFN + j] + bhh[DFFN + j]);
        float gg = tanhf(g4[2] + bih[2 * DFFN + j] + bhh[2 * DFFN + j]);
        float go = sigm(g4[3] + bih[3 * DFFN + j] + bhh[3 * DFFN + j]);
        float c = gf * cin[(size_t)b * DFFN + j] + gi * gg;
        float h = go * tanhf(c);
        cout[(size_t)b * DFFN + j] = c;
        hout[(size_t)b * DFFN + j] = h;
        ys[((size_t)b * 64 + t) * DFFN + j] = h;
    }
}

// ---------------------------------------------------------------- host orchestration
struct AttnW {
    const u16 *wq, *wk, *wv, *wo, *w1, *w2;
    const float *bq, *bk, *bv, *bo, *b1, *b2, *ln1g, *ln1b, *ln2g, *ln2b;
};

extern "C" void kernel_launch(void* const* d_in, const int* in_sizes, int n_in,
                              void* d_out, int out_size, void* d_ws, size_t ws_size,
                              hipStream_t stream) {
    (void)in_sizes; (void)n_in; (void)out_size; (void)ws_size;
    auto cdiv = [](long long a, long long b) { return (int)((a + b - 1) / b); };

    // ---- workspace bump allocator
    char* wsb = (char*)d_ws;
    size_t off = 0;
    auto alloc = [&](size_t bytes) -> void* {
        void* p = wsb + off;
        off += (bytes + 255) & ~(size_t)255;
        return p;
    };
    auto cvt = [&](int idx, size_t n) -> const u16* {
        u16* dst = (u16*)alloc(n * sizeof(u16));
        k_cvt<<<cdiv((long long)n, 256), 256, 0, stream>>>((const float*)d_in[idx], dst, (int)n);
        return dst;
    };

    // ---- input indices (flattened setup_inputs order)
    // 0:x 1:emb | prob_layers i: base 2+16i | rpr: 50 | conv i: 66+4i | enc_norm: 86,87 | lstm: 88..95 | wout1: 96,97
    auto mkattn = [&](int base) {
        AttnW w;
        w.wq = cvt(base + 0, (size_t)DMODEL * DMODEL);  w.bq = (const float*)d_in[base + 1];
        w.wk = cvt(base + 2, (size_t)DMODEL * DMODEL);  w.bk = (const float*)d_in[base + 3];
        w.wv = cvt(base + 4, (size_t)DMODEL * DMODEL);  w.bv = (const float*)d_in[base + 5];
        w.wo = cvt(base + 6, (size_t)DMODEL * DMODEL);  w.bo = (const float*)d_in[base + 7];
        w.w1 = cvt(base + 8, (size_t)DFFN * DMODEL);    w.b1 = (const float*)d_in[base + 9];
        w.w2 = cvt(base + 10, (size_t)DMODEL * DFFN);   w.b2 = (const float*)d_in[base + 11];
        w.ln1g = (const float*)d_in[base + 12]; w.ln1b = (const float*)d_in[base + 13];
        w.ln2g = (const float*)d_in[base + 14]; w.ln2b = (const float*)d_in[base + 15];
        return w;
    };
    AttnW probw[3] = { mkattn(2), mkattn(18), mkattn(34) };
    AttnW rprw = mkattn(50);
    const u16* convw[5];
    for (int i = 0; i < 5; i++) convw[i] = cvt(66 + 4 * i, (size_t)DMODEL * DMODEL * 3);
    const float* encg = (const float*)d_in[86];
    const float* encb = (const float*)d_in[87];
    const u16* wih0 = cvt(88, (size_t)4 * DFFN * DMODEL);
    const u16* whh0 = cvt(89, (size_t)4 * DFFN * DFFN);
    const float* bih0 = (const float*)d_in[90]; const float* bhh0 = (const float*)d_in[91];
    const u16* wih1 = cvt(92, (size_t)4 * DFFN * DFFN);
    const u16* whh1 = cvt(93, (size_t)4 * DFFN * DFFN);
    const float* bih1 = (const float*)d_in[94]; const float* bhh1 = (const float*)d_in[95];
    const u16* woutw = cvt(96, (size_t)NVOCAB * DFFN);
    const float* woutb = (const float*)d_in[97];

    // ---- activation buffers
    size_t ntok = (size_t)NB * SEQLEN;
    float* h_f = (float*)alloc(ntok * DMODEL * 4);
    u16*   h_b = (u16*)alloc(ntok * DMODEL * 2);
    u16*   qb  = (u16*)alloc(ntok * DMODEL * 2);
    u16*   kb  = (u16*)alloc(ntok * DMODEL * 2);
    u16*   vb  = (u16*)alloc(ntok * DMODEL * 2);
    u16*   ctxb = (u16*)alloc(ntok * DMODEL * 2);
    float* tmpF = (float*)alloc(ntok * DMODEL * 4);
    u16*   bigB = (u16*)alloc(ntok * 3 * DMODEL * 2);   // im2col (K=1536) & ffn hidden (K=1024)
    float* Mbuf = (float*)alloc((size_t)NB * NHEAD * SEQLEN * 4);
    int*   mtop = (int*)alloc((size_t)NB * NHEAD * 256 * 4);
    float* ys0  = (float*)alloc((size_t)NB * 64 * DFFN * 4);
    float* ys1  = (float*)alloc((size_t)NB * 64 * DFFN * 4);
    u16*   ys1b = (u16*)alloc((size_t)NB * 64 * DFFN * 2);
    float* hA = (float*)alloc((size_t)NB * DFFN * 4);
    float* hB = (float*)alloc((size_t)NB * DFFN * 4);
    float* cA = (float*)alloc((size_t)NB * DFFN * 4);
    float* cB = (float*)alloc((size_t)NB * DFFN * 4);

    auto gemm = [&](const u16* A, const u16* W, const float* bias, float* oF, u16* oB,
                    int M, int N, int K, int relu) {
        dim3 g(cdiv(N, TN), cdiv(M, TM));
        if ((M % TM) == 0 && (N % TN) == 0)
            k_gemm<false><<<g, 256, 0, stream>>>(A, W, bias, oF, oB, M, N, K, relu);
        else
            k_gemm<true><<<g, 256, 0, stream>>>(A, W, bias, oF, oB, M, N, K, relu);
    };
    auto ufor = [](int L) { int u = (int)(30.0 * std::ceil(std::log((double)L))); return u < L ? u : L; };

    // ---- embedding + PE
    k_embed<<<cdiv(ntok * DMODEL, 256), 256, 0, stream>>>((const int*)d_in[0], (const float*)d_in[1], h_f, h_b);

    auto attn_layer = [&](const AttnW& w, int L, bool prob) {
        int M = NB * L;
        gemm(h_b, w.wq, w.bq, nullptr, qb, M, DMODEL, DMODEL, 0);
        gemm(h_b, w.wk, w.bk, nullptr, kb, M, DMODEL, DMODEL, 0);
        gemm(h_b, w.wv, w.bv, nullptr, vb, M, DMODEL, DMODEL, 0);
        if (prob) {
            int u = ufor(L);
            k_probM<<<cdiv((long long)NB * NHEAD * L, 256), 256, 0, stream>>>(qb, kb, Mbuf, L, u);
            k_topu<<<NB * NHEAD, 256, 0, stream>>>(Mbuf, mtop, L, u);
            k_vmean<<<NB * NHEAD, HDIM, 0, stream>>>(vb, ctxb, L);
            k_toprows<<<NB * NHEAD * u, 256, 0, stream>>>(qb, kb, vb, mtop, ctxb, L, u);
        } else {
            k_causal<<<NB * NHEAD * L, 256, 0, stream>>>(qb, kb, vb, ctxb, L);
        }
        gemm(ctxb, w.wo, w.bo, tmpF, nullptr, M, DMODEL, DMODEL, 0);
        k_ln<<<M, 256, 0, stream>>>(tmpF, h_f, w.ln1g, w.ln1b, h_f, h_b, M);
        gemm(h_b, w.w1, w.b1, nullptr, bigB, M, DFFN, DMODEL, 1);     // fused ReLU
        gemm(bigB, w.w2, w.b2, tmpF, nullptr, M, DMODEL, DFFN, 0);
        k_ln<<<M, 256, 0, stream>>>(tmpF, h_f, w.ln2g, w.ln2b, h_f, h_b, M);
    };
    auto conv_layer = [&](int ci, int L) -> int {
        int M = NB * L;
        k_im2col<<<cdiv((long long)M * DMODEL, 256), 256, 0, stream>>>(h_b, bigB, L);
        gemm(bigB, convw[ci], (const float*)d_in[67 + 4 * ci], tmpF, nullptr, M, DMODEL, 3 * DMODEL, 0);
        int Lo = (L - 1) / 2 + 1;
        k_pool<<<cdiv((long long)NB * Lo * DMODEL, 256), 256, 0, stream>>>(
            tmpF, (const float*)d_in[68 + 4 * ci], (const float*)d_in[69 + 4 * ci], h_f, h_b, L, Lo);
        return Lo;
    };

    int L = SEQLEN;
    for (int i = 0; i < 3; i++) {            // Informer prob-attention stack with distilling
        attn_layer(probw[i], L, true);
        L = conv_layer(i, L);                // 2048 -> 1024 -> 512 -> 256
    }
    for (int j = 0; j < 3; j++) {            // shared-weight causal encoder stack
        attn_layer(rprw, L, false);
        if (j < 2) L = conv_layer(3 + j, L); // 256 -> 128 -> 64
    }
    // final encoder norm  (L == 64)
    k_ln<<<NB * L, 256, 0, stream>>>(h_f, nullptr, encg, encb, h_f, h_b, NB * L);

    // ---- 2-layer LSTM over T=64
    int nhc = NB * DFFN;
    k_fill0<<<cdiv(nhc, 256), 256, 0, stream>>>(hA, nhc);
    k_fill0<<<cdiv(nhc, 256), 256, 0, stream>>>(cA, nhc);
    float *hc = hA, *hn = hB, *cc = cA, *cn = cB;
    for (int t = 0; t < 64; t++) {
        k_lstm_step<<<NB * DFFN / 8, 256, 0, stream>>>(h_f, wih0, whh0, bih0, bhh0, hc, cc, hn, cn, ys0, t, DMODEL);
        float* ts;
        ts = hc; hc = hn; hn = ts;
        ts = cc; cc = cn; cn = ts;
    }
    k_fill0<<<cdiv(nhc, 256), 256, 0, stream>>>(hc, nhc);
    k_fill0<<<cdiv(nhc, 256), 256, 0, stream>>>(cc, nhc);
    for (int t = 0; t < 64; t++) {
        k_lstm_step<<<NB * DFFN / 8, 256, 0, stream>>>(ys0, wih1, whh1, bih1, bhh1, hc, cc, hn, cn, ys1, t, DFFN);
        float* ts;
        ts = hc; hc = hn; hn = ts;
        ts = cc; cc = cn; cn = ts;
    }

    // ---- output projection  [NB*64, 1024] x [390, 1024]^T -> d_out (f32)
    k_cvt<<<cdiv((long long)NB * 64 * DFFN, 256), 256, 0, stream>>>(ys1, ys1b, NB * 64 * DFFN);
    gemm(ys1b, woutw, woutb, (float*)d_out, nullptr, NB * 64, NVOCAB, DFFN, 0);
}
